// ODEfunc_40175124087283
// MI455X (gfx1250) — matrix operand order repaired and checked by reading, not compiled, not run on hardware
//
#include <hip/hip_runtime.h>

#define Bsz 16384
#define Dd 256
#define Hh 256
#define Ss 32

typedef __attribute__((ext_vector_type(16))) __bf16 bf16x16;
typedef __attribute__((ext_vector_type(8)))  float  f32x8;
typedef __attribute__((ext_vector_type(4)))  unsigned int u32x4;
typedef __attribute__((ext_vector_type(8)))  int  i32x8;
typedef __attribute__((ext_vector_type(4)))  int  i32x4;

__device__ __forceinline__ f32x8 zero8() {
  f32x8 z;
#pragma unroll
  for (int i = 0; i < 8; ++i) z[i] = 0.0f;
  return z;
}

// B fragment (32x16): lane l -> col = l%16, K = (l/16)*16 + idx, 16 contiguous bf16.
__device__ __forceinline__ bf16x16 load_b_frag(const __bf16* p) {
  union { bf16x16 v; uint4 u[2]; } t;
  t.u[0] = ((const uint4*)p)[0];
  t.u[1] = ((const uint4*)p)[1];
  return t.v;
}

// A fragment (16x32) from bf16 LDS: 16 contiguous elements {base..base+15}.
__device__ __forceinline__ bf16x16 load_a_frag(const __bf16* p) {
  union { bf16x16 v; uint4 u[2]; } t;
  t.u[0] = *(const uint4*)p;
  t.u[1] = *(const uint4*)(p + 8);
  return t.v;
}

// A fragment from f32 LDS (TDM-staged, convert at read): 16 contiguous f32.
__device__ __forceinline__ bf16x16 load_a_frag_f32(const float* p) {
  float4 a0 = ((const float4*)p)[0];
  float4 a1 = ((const float4*)p)[1];
  float4 b0 = ((const float4*)(p + 8))[0];
  float4 b1 = ((const float4*)(p + 8))[1];
  bf16x16 r;
  r[0] = (__bf16)a0.x; r[1] = (__bf16)a0.y; r[2]  = (__bf16)a0.z; r[3]  = (__bf16)a0.w;
  r[4] = (__bf16)a1.x; r[5] = (__bf16)a1.y; r[6]  = (__bf16)a1.z; r[7]  = (__bf16)a1.w;
  r[8] = (__bf16)b0.x; r[9] = (__bf16)b0.y; r[10] = (__bf16)b0.z; r[11] = (__bf16)b0.w;
  r[12] = (__bf16)b1.x; r[13] = (__bf16)b1.y; r[14] = (__bf16)b1.z; r[15] = (__bf16)b1.w;
  return r;
}

// TDM: DMA a 32-row x 256-col f32 tile (row stride 256) from global into LDS.
__device__ __forceinline__ void tdm_load_tile(unsigned lds_off, const float* gptr) {
  unsigned long long ga = (unsigned long long)(size_t)gptr;
  u32x4 g0;
  g0[0] = 1u;                                   // count=1, user descriptor
  g0[1] = lds_off;                              // lds_addr (bytes)
  g0[2] = (unsigned)ga;                         // global_addr[31:0]
  g0[3] = (unsigned)(ga >> 32) | (2u << 30);    // global_addr[56:32] | type=2
  i32x8 g1;
  g1[0] = (int)(2u << 16);                      // data_size = 4B; wg_mask=0
  g1[1] = (int)(256u << 16);                    // tensor_dim0 = 256 (bits 79:48)
  g1[2] = (int)(32u << 16);                     // tensor_dim1 = 32  (bits 111:80)
  g1[3] = (int)(256u << 16);                    // tile_dim0 = 256   (bits 127:112)
  g1[4] = 32;                                   // tile_dim1 = 32    (bits 143:128)
  g1[5] = 256;                                  // tensor_dim0_stride = 256
  g1[6] = 0;
  g1[7] = 0;
  i32x4 z4;
  z4[0] = 0; z4[1] = 0; z4[2] = 0; z4[3] = 0;   // 2D tile: groups 2/3 unused
  i32x8 z8;
#pragma unroll
  for (int i = 0; i < 8; ++i) z8[i] = 0;
  __builtin_amdgcn_tensor_load_to_lds(g0, g1, z4, z4, z8, 0);
}

// ---------------- Kernel 0: weight conversion to bf16 ----------------
__global__ __launch_bounds__(256) void convert_weights(
    const float* __restrict__ W1, const float* __restrict__ W2,
    __bf16* __restrict__ W1T, __bf16* __restrict__ W2b, __bf16* __restrict__ W2T) {
  const int i = blockIdx.x * 256 + threadIdx.x;   // 0 .. 65535
  const int d1 = i / Hh, h1 = i % Hh;             // W1 is [D][H]
  W1T[h1 * Dd + d1] = (__bf16)W1[i];
  const int h2 = i / Dd, d2 = i % Dd;             // W2 is [H][D]
  const float w2 = W2[i];
  W2b[i] = (__bf16)w2;
  W2T[d2 * Hh + h2] = (__bf16)w2;
}

// ---------------- Kernel 1: dy = tanh(y@W1 + b1 + t*wt) @ W2 + b2, dbuf = 1 - a^2 ----
__global__ __launch_bounds__(256) void fwd_kernel(
    const float* __restrict__ y, const float* __restrict__ b1,
    const float* __restrict__ wt, const float* __restrict__ b2,
    const float* __restrict__ tt, const __bf16* __restrict__ W1T,
    const __bf16* __restrict__ W2T, float* __restrict__ dy,
    float* __restrict__ dbuf) {
  __shared__ __bf16 ylds[32 * Dd];
  __shared__ __bf16 alds[32 * Hh];
  const int tid = threadIdx.x;
  const int wv = tid >> 5, ln = tid & 31;
  const int half = ln >> 4, lc = ln & 15;
  const int rowbase = blockIdx.x * 32;

  {
    const float4* src = (const float4*)(y + (size_t)rowbase * Dd);
#pragma unroll
    for (int k = tid; k < 32 * Dd / 4; k += 256) {
      float4 v = src[k];
      __bf16* dst = &ylds[k * 4];
      dst[0] = (__bf16)v.x; dst[1] = (__bf16)v.y;
      dst[2] = (__bf16)v.z; dst[3] = (__bf16)v.w;
    }
  }
  __syncthreads();
  const float t0 = tt[0];

  // ---- GEMM 1: h = y @ W1 ----
  f32x8 hacc[2][2];
#pragma unroll
  for (int mt = 0; mt < 2; ++mt)
#pragma unroll
    for (int j = 0; j < 2; ++j) hacc[mt][j] = zero8();

  for (int kt = 0; kt < 8; ++kt) {
    bf16x16 Bf[2];
#pragma unroll
    for (int j = 0; j < 2; ++j) {
      const int h = (2 * wv + j) * 16 + lc;
      Bf[j] = load_b_frag(&W1T[h * Dd + kt * 32 + half * 16]);
    }
#pragma unroll
    for (int mt = 0; mt < 2; ++mt) {
      bf16x16 A = load_a_frag(&ylds[(mt * 16 + lc) * Dd + kt * 32 + half * 16]);
#pragma unroll
      for (int j = 0; j < 2; ++j)
        hacc[mt][j] = __builtin_amdgcn_wmma_f32_16x16x32_bf16(
            false, A, false, Bf[j], (short)0, hacc[mt][j], false, false);
    }
  }

  // bias + tanh; d -> global, a -> LDS (bf16)
#pragma unroll
  for (int j = 0; j < 2; ++j) {
    const int h = (2 * wv + j) * 16 + lc;
    const float bias = b1[h] + t0 * wt[h];
#pragma unroll
    for (int mt = 0; mt < 2; ++mt) {
#pragma unroll
      for (int r = 0; r < 8; ++r) {
        const int row = mt * 16 + half * 8 + r;
        const float a = tanhf(hacc[mt][j][r] + bias);
        dbuf[(size_t)(rowbase + row) * Hh + h] = 1.0f - a * a;
        alds[row * Hh + h] = (__bf16)a;
      }
    }
  }
  __syncthreads();

  // ---- GEMM 2: dy = a @ W2 + b2 ----
  f32x8 oacc[2][2];
#pragma unroll
  for (int mt = 0; mt < 2; ++mt)
#pragma unroll
    for (int j = 0; j < 2; ++j) oacc[mt][j] = zero8();

  for (int kt = 0; kt < 8; ++kt) {
    bf16x16 Bf[2];
#pragma unroll
    for (int j = 0; j < 2; ++j) {
      const int dcol = (2 * wv + j) * 16 + lc;
      Bf[j] = load_b_frag(&W2T[dcol * Hh + kt * 32 + half * 16]);
    }
#pragma unroll
    for (int mt = 0; mt < 2; ++mt) {
      bf16x16 A = load_a_frag(&alds[(mt * 16 + lc) * Hh + kt * 32 + half * 16]);
#pragma unroll
      for (int j = 0; j < 2; ++j)
        oacc[mt][j] = __builtin_amdgcn_wmma_f32_16x16x32_bf16(
            false, A, false, Bf[j], (short)0, oacc[mt][j], false, false);
    }
  }

#pragma unroll
  for (int j = 0; j < 2; ++j) {
    const int dcol = (2 * wv + j) * 16 + lc;
    const float bb = b2[dcol];
#pragma unroll
    for (int mt = 0; mt < 2; ++mt)
#pragma unroll
      for (int r = 0; r < 8; ++r)
        dy[(size_t)(rowbase + mt * 16 + half * 8 + r) * Dd + dcol] =
            oacc[mt][j][r] + bb;
  }
}

// ---------------- Kernel 2: div via TDM double-buffered e tiles ----------------
__global__ __launch_bounds__(256) void div_kernel(
    const float* __restrict__ e, const __bf16* __restrict__ W1T,
    const __bf16* __restrict__ W2b, const float* __restrict__ dbuf,
    float* __restrict__ outdiv) {
  __shared__ float ebuf[2][32 * Dd];    // 2 x 32KB TDM-staged f32 tiles
  __shared__ float divl[32];
  const int tid = threadIdx.x;
  const int wv = tid >> 5, ln = tid & 31;
  const int half = ln >> 4, lc = ln & 15;
  const int rowbase = blockIdx.x * 32;

  if (tid < 32) divl[tid] = 0.0f;

  const unsigned lds_base = (unsigned)(size_t)(void*)&ebuf[0][0];

  // Preload d (tanh') into registers matching the C/D tile layout.
  float dreg[2][2][8];
#pragma unroll
  for (int mt = 0; mt < 2; ++mt)
#pragma unroll
    for (int j = 0; j < 2; ++j) {
      const int h = (2 * wv + j) * 16 + lc;
#pragma unroll
      for (int r = 0; r < 8; ++r)
        dreg[mt][j][r] =
            dbuf[(size_t)(rowbase + mt * 16 + half * 8 + r) * Hh + h];
    }

  float part[2][2][8];
#pragma unroll
  for (int mt = 0; mt < 2; ++mt)
#pragma unroll
    for (int j = 0; j < 2; ++j)
#pragma unroll
      for (int r = 0; r < 8; ++r) part[mt][j][r] = 0.0f;

  // Prologue: DMA tile 0 into buffer 0 (wave 0 issues; TENSORcnt tracks).
  if (wv == 0)
    tdm_load_tile(lds_base, e + (size_t)rowbase * Dd);

  for (int s = 0; s < Ss; ++s) {
    if (wv == 0) {
      if (s + 1 < Ss) {
        // Prefetch next tile into the other buffer, then wait until only
        // that one is outstanding (=> tile s has landed; TDM is in-order).
        tdm_load_tile(lds_base + (unsigned)(((s + 1) & 1) * 32 * Dd * 4),
                      e + ((size_t)(s + 1) * Bsz + rowbase) * Dd);
        __builtin_amdgcn_s_wait_tensorcnt(1);
      } else {
        __builtin_amdgcn_s_wait_tensorcnt(0);
      }
    }
    __syncthreads();   // tile s visible to all waves

    const float* ebase = &ebuf[s & 1][0];
    f32x8 u[2][2], v[2][2];
#pragma unroll
    for (int mt = 0; mt < 2; ++mt)
#pragma unroll
      for (int j = 0; j < 2; ++j) { u[mt][j] = zero8(); v[mt][j] = zero8(); }

    for (int kt = 0; kt < 8; ++kt) {
      bf16x16 Bu[2], Bv[2];
#pragma unroll
      for (int j = 0; j < 2; ++j) {
        const int h = (2 * wv + j) * 16 + lc;
        Bu[j] = load_b_frag(&W1T[h * Dd + kt * 32 + half * 16]);
        Bv[j] = load_b_frag(&W2b[h * Dd + kt * 32 + half * 16]);
      }
#pragma unroll
      for (int mt = 0; mt < 2; ++mt) {
        bf16x16 A = load_a_frag_f32(&ebase[(mt * 16 + lc) * Dd + kt * 32 + half * 16]);
#pragma unroll
        for (int j = 0; j < 2; ++j) {
          u[mt][j] = __builtin_amdgcn_wmma_f32_16x16x32_bf16(
              false, A, false, Bu[j], (short)0, u[mt][j], false, false);
          v[mt][j] = __builtin_amdgcn_wmma_f32_16x16x32_bf16(
              false, A, false, Bv[j], (short)0, v[mt][j], false, false);
        }
      }
    }
#pragma unroll
    for (int mt = 0; mt < 2; ++mt)
#pragma unroll
      for (int j = 0; j < 2; ++j)
#pragma unroll
        for (int r = 0; r < 8; ++r)
          part[mt][j][r] += dreg[mt][j][r] * u[mt][j][r] * v[mt][j][r];

    __syncthreads();   // all reads of ebuf[s&1] done before it's re-filled at s+2
  }

  // Reduce across the 16 columns held by lanes {0..15} / {16..31}.
#pragma unroll
  for (int mt = 0; mt < 2; ++mt) {
#pragma unroll
    for (int r = 0; r < 8; ++r) {
      float vsum = part[mt][0][r] + part[mt][1][r];
      vsum += __shfl_xor(vsum, 1, 32);
      vsum += __shfl_xor(vsum, 2, 32);
      vsum += __shfl_xor(vsum, 4, 32);
      vsum += __shfl_xor(vsum, 8, 32);
      if (lc == 0) atomicAdd(&divl[mt * 16 + half * 8 + r], vsum);
    }
  }
  __syncthreads();
  if (tid < 32) outdiv[rowbase + tid] = -divl[tid] * (1.0f / Ss);
}

// ---------------- Host launcher ----------------
extern "C" void kernel_launch(void* const* d_in, const int* in_sizes, int n_in,
                              void* d_out, int out_size, void* d_ws, size_t ws_size,
                              hipStream_t stream) {
  const float* t  = (const float*)d_in[0];
  const float* y  = (const float*)d_in[1];
  const float* e  = (const float*)d_in[2];
  const float* W1 = (const float*)d_in[3];
  const float* b1 = (const float*)d_in[4];
  const float* wt = (const float*)d_in[5];
  const float* W2 = (const float*)d_in[6];
  const float* b2 = (const float*)d_in[7];

  float* dy   = (float*)d_out;                    // [B, D]
  float* ndiv = dy + (size_t)Bsz * Dd;            // [B, 1] = -div

  float*  dbuf = (float*)d_ws;                                    // B*H f32
  __bf16* W1T  = (__bf16*)((char*)d_ws + (size_t)Bsz * Hh * 4);   // H*D bf16
  __bf16* W2b  = W1T + (size_t)Dd * Hh;                           // H*D bf16
  __bf16* W2T  = W2b + (size_t)Dd * Hh;                           // D*H bf16

  convert_weights<<<(Dd * Hh) / 256, 256, 0, stream>>>(W1, W2, W1T, W2b, W2T);
  fwd_kernel<<<Bsz / 32, 256, 0, stream>>>(y, b1, wt, b2, t, W1T, W2T, dy, dbuf);
  div_kernel<<<Bsz / 32, 256, 0, stream>>>(e, W1T, W2b, dbuf, ndiv);
}
